// EEGLstmNet_82N_200_59760174956974
// MI455X (gfx1250) — matrix operand order repaired and checked by reading, not compiled, not run on hardware
//
#include <hip/hip_runtime.h>
#include <hip/hip_bf16.h>

typedef __attribute__((ext_vector_type(16))) __bf16 v16bf;
typedef __attribute__((ext_vector_type(8)))  __bf16 v8bf;
typedef __attribute__((ext_vector_type(8)))  float  v8f;

union AFrag { v16bf v; v8bf h[2]; unsigned short s[16]; };

__device__ __forceinline__ unsigned short f2bf(float f) {
    unsigned int u = __float_as_uint(f);
    u += 0x7FFFu + ((u >> 16) & 1u);          // round-to-nearest-even
    return (unsigned short)(u >> 16);
}
__device__ __forceinline__ float bf2f(unsigned short s) {
    return __uint_as_float(((unsigned int)s) << 16);
}
__device__ __forceinline__ float eluf(float x) { return x > 0.f ? x : (__expf(x) - 1.f); }
__device__ __forceinline__ float sigmoidf(float x) { return 1.f / (1.f + __expf(-x)); }

// ---------------------------------------------------------------------------
// conv1: [B,1,200,64] * w[8,1,101,1], pad (50,50)  -> ELU -> store + BN1 stats
// Implicit GEMM on WMMA bf16: A = weights [16(ch)x128(k)], B = shifted input
// windows [128 x 16 h-positions] per (b, w) column.
// grid: (1024, 4) blocks of 256 (8 waves); block = one b, 16 w columns.
// ---------------------------------------------------------------------------
__global__ __launch_bounds__(256) void k_conv1(const float* __restrict__ x,
                                               const float* __restrict__ w1,
                                               const float* __restrict__ b1,
                                               float* __restrict__ out,
                                               float* __restrict__ stats1) {
    __shared__ __align__(16) unsigned short xcol[16][336];
    __shared__ __align__(16) unsigned short wTs[16][128];
    __shared__ float ssum[8], ssq[8];

    const int b   = blockIdx.x;
    const int w0  = blockIdx.y * 16;
    const int tid = threadIdx.x;
    if (tid < 8) { ssum[tid] = 0.f; ssq[tid] = 0.f; }

    for (int e = tid; e < 16 * 336; e += 256) {
        int wc = e / 336, i = e % 336;
        int hh = i - 50;
        float v = (hh >= 0 && hh < 200) ? x[(b * 200 + hh) * 64 + (w0 + wc)] : 0.f;
        xcol[wc][i] = f2bf(v);
    }
    for (int e = tid; e < 16 * 128; e += 256) {
        int c = e >> 7, k = e & 127;
        float v = (c < 8 && k < 101) ? w1[c * 101 + k] : 0.f;
        wTs[c][k] = f2bf(v);
    }
    __syncthreads();

    const int lane = tid & 31, wv = tid >> 5;
    const int half = lane >> 4, ln = lane & 15;

    AFrag afr[4];
#pragma unroll
    for (int ks = 0; ks < 4; ++ks) {
        int kb = ks * 32 + half * 16;
#pragma unroll
        for (int e = 0; e < 8; ++e) {
            afr[ks].s[e]     = wTs[ln][kb + e];
            afr[ks].s[8 + e] = wTs[ln][kb + 8 + e];
        }
    }

    float ls[8], lq[8];
#pragma unroll
    for (int r = 0; r < 8; ++r) { ls[r] = 0.f; lq[r] = 0.f; }

    for (int wl = 0; wl < 2; ++wl) {
        int wc = wv * 2 + wl;
        for (int ht = 0; ht < 13; ++ht) {
            v8f acc = {};
#pragma unroll
            for (int ks = 0; ks < 4; ++ks) {
                AFrag bfr;
                int base = ht * 16 + ln + half * 16 + ks * 32;
#pragma unroll
                for (int e = 0; e < 16; ++e) bfr.s[e] = xcol[wc][base + e];
                acc = __builtin_amdgcn_wmma_f32_16x16x32_bf16(
                    false, afr[ks].v, false, bfr.v, (short)0, acc, false, false);
            }
            int h = ht * 16 + ln;
            if (half == 0 && h < 200) {
#pragma unroll
                for (int r = 0; r < 8; ++r) {
                    float v = eluf(acc[r] + b1[r]);
                    out[((b * 8 + r) * 200 + h) * 64 + w0 + wc] = v;
                    ls[r] += v; lq[r] += v * v;
                }
            }
        }
    }
    if (half == 0) {
#pragma unroll
        for (int r = 0; r < 8; ++r) { atomicAdd(&ssum[r], ls[r]); atomicAdd(&ssq[r], lq[r]); }
    }
    __syncthreads();
    if (tid < 8) { atomicAdd(&stats1[tid], ssum[tid]); atomicAdd(&stats1[8 + tid], ssq[tid]); }
}

// stats layout: [sum nch][sumsq nch][mean nch][inv_std nch]
__global__ void k_finalize(float* stats, int nch, float invN) {
    int c = threadIdx.x;
    if (c < nch) {
        float m = stats[c] * invN;
        float v = stats[nch + c] * invN - m * m;
        stats[2 * nch + c] = m;
        stats[3 * nch + c] = rsqrtf(v);
    }
}

// ---------------------------------------------------------------------------
// conv2 (groups=4, kernel 1x64) over BN1(conv1 out) -> ELU -> [B,16,200] + BN2 stats
// block: 16 (b,h) pairs x 16 channels
// ---------------------------------------------------------------------------
__global__ __launch_bounds__(256) void k_conv2(const float* __restrict__ c1,
                                               const float* __restrict__ stats1,
                                               const float* __restrict__ bn1w,
                                               const float* __restrict__ bn1b,
                                               const float* __restrict__ w2,
                                               const float* __restrict__ b2,
                                               float* __restrict__ c2,
                                               float* __restrict__ stats2) {
    __shared__ float in0[16][512];
    __shared__ float ws[2048];
    __shared__ float ssum[16], ssq[16];
    const int tid = threadIdx.x;
    if (tid < 16) { ssum[tid] = 0.f; ssq[tid] = 0.f; }
    for (int e = tid; e < 2048; e += 256) ws[e] = w2[e];

    const int p0 = blockIdx.x * 16;
    for (int e = tid; e < 16 * 512; e += 256) {
        int pl = e >> 9, idx = e & 511;
        int ci = idx >> 6, wq = idx & 63;
        int p = p0 + pl, bb = p / 200, hh = p % 200;
        float v = c1[((bb * 8 + ci) * 200 + hh) * 64 + wq];
        in0[pl][idx] = (v - stats1[16 + ci]) * stats1[24 + ci] * bn1w[ci] + bn1b[ci];
    }
    __syncthreads();

    int pl = tid >> 4, c = tid & 15, g = c >> 2;
    float acc = b2[c];
    const float* wrow = &ws[c * 128];
    const float* irow = &in0[pl][g * 128];
#pragma unroll 8
    for (int q = 0; q < 128; ++q) acc += wrow[q] * irow[q];
    acc = eluf(acc);
    int p = p0 + pl, bb = p / 200, hh = p % 200;
    c2[(bb * 16 + c) * 200 + hh] = acc;
    atomicAdd(&ssum[c], acc); atomicAdd(&ssq[c], acc * acc);
    __syncthreads();
    if (tid < 16) { atomicAdd(&stats2[tid], ssum[tid]); atomicAdd(&stats2[16 + tid], ssq[tid]); }
}

// BN2 apply + maxpool(4) -> [B,16,50]
__global__ void k_pool4(const float* __restrict__ c2, const float* __restrict__ stats2,
                        const float* __restrict__ bn2w, const float* __restrict__ bn2b,
                        float* __restrict__ p) {
    int i = blockIdx.x * 256 + threadIdx.x;
    if (i >= 1024 * 16 * 50) return;
    int hp = i % 50, c = (i / 50) & 15, bb = i / (50 * 16);
    float a = stats2[48 + c] * bn2w[c];
    float bc = bn2b[c] - stats2[32 + c] * a;
    const float* s = &c2[(bb * 16 + c) * 200 + hp * 4];
    float v = s[0] * a + bc;
    v = fmaxf(v, s[1] * a + bc);
    v = fmaxf(v, s[2] * a + bc);
    v = fmaxf(v, s[3] * a + bc);
    p[(bb * 16 + c) * 50 + hp] = v;
}

// conv3 (groups=8, 25x1, pad 12) -> ELU -> convp (1x1 full) -> ELU + BN3 stats
// block: one b, all channels/heights in LDS
__global__ __launch_bounds__(256) void k_conv3p(const float* __restrict__ p,
                                                const float* __restrict__ w3,
                                                const float* __restrict__ b3,
                                                const float* __restrict__ wp,
                                                const float* __restrict__ bp,
                                                float* __restrict__ t4,
                                                float* __restrict__ stats3) {
    __shared__ float pin[16][50];
    __shared__ float t3s[16][50];
    __shared__ float ssum[16], ssq[16];
    const int bb = blockIdx.x, tid = threadIdx.x;
    if (tid < 16) { ssum[tid] = 0.f; ssq[tid] = 0.f; }
    for (int e = tid; e < 800; e += 256)
        pin[e / 50][e % 50] = p[(bb * 16 + e / 50) * 50 + e % 50];
    __syncthreads();
    for (int e = tid; e < 800; e += 256) {
        int c = e / 50, h = e % 50, g2 = c >> 1;
        float acc = b3[c];
        for (int ci = 0; ci < 2; ++ci) {
            const float* wr = &w3[(c * 2 + ci) * 25];
            const float* xr = pin[2 * g2 + ci];
#pragma unroll
            for (int k = 0; k < 25; ++k) {
                int hh = h + k - 12;
                if (hh >= 0 && hh < 50) acc += wr[k] * xr[hh];
            }
        }
        t3s[c][h] = eluf(acc);
    }
    __syncthreads();
    for (int e = tid; e < 800; e += 256) {
        int c = e / 50, h = e % 50;
        float acc = bp[c];
#pragma unroll
        for (int ci = 0; ci < 16; ++ci) acc += wp[c * 16 + ci] * t3s[ci][h];
        acc = eluf(acc);
        t4[(bb * 16 + c) * 50 + h] = acc;
        atomicAdd(&ssum[c], acc); atomicAdd(&ssq[c], acc * acc);
    }
    __syncthreads();
    if (tid < 16) { atomicAdd(&stats3[tid], ssum[tid]); atomicAdd(&stats3[16 + tid], ssq[tid]); }
}

// BN3 apply + maxpool(2) -> x25 [B,16,25]
__global__ void k_pool2(const float* __restrict__ t4, const float* __restrict__ stats3,
                        const float* __restrict__ bn3w, const float* __restrict__ bn3b,
                        float* __restrict__ x25) {
    int i = blockIdx.x * 256 + threadIdx.x;
    if (i >= 1024 * 16 * 25) return;
    int t = i % 25, c = (i / 25) & 15, bb = i / (25 * 16);
    float a = stats3[48 + c] * bn3w[c];
    float bc = bn3b[c] - stats3[32 + c] * a;
    const float* s = &t4[(bb * 16 + c) * 50 + t * 2];
    x25[(bb * 16 + c) * 25 + t] = fmaxf(s[0] * a + bc, s[1] * a + bc);
}

// Backward LSTM direction: only hs[:,1,0] is consumed = one step from zero state.
__global__ void k_lstm_bwd(const float* __restrict__ x25, const float* __restrict__ wih,
                           const float* __restrict__ bih, const float* __restrict__ bhh,
                           float* __restrict__ feat) {
    int i = blockIdx.x * 256 + threadIdx.x;
    if (i >= 16 * 1024 * 50) return;
    int j = i % 50, bb = (i / 50) & 1023, c = i / (50 * 1024);
    float xv = x25[(bb * 16 + c) * 25 + 24];
    int base = (c * 2 + 1) * 200;
    float gi = xv * wih[base + j]       + bih[base + j]       + bhh[base + j];
    float gg = xv * wih[base + 100 + j] + bih[base + 100 + j] + bhh[base + 100 + j];
    float go = xv * wih[base + 150 + j] + bih[base + 150 + j] + bhh[base + 150 + j];
    float cc = sigmoidf(gi) * tanhf(gg);
    feat[(c * 1024 + bb) * 100 + 50 + j] = sigmoidf(go) * tanhf(cc);
}

// ---------------------------------------------------------------------------
// Forward LSTM recurrence with WMMA: per step g = h @ whh^T (M=32 batch rows,
// N=200 gates, K=50->64 padded), bf16 16x16x32 tiles; then gate VALU phase.
// grid: 16 channels x 32 batch tiles = 512 blocks of 256 (8 waves).
// ---------------------------------------------------------------------------
#define BT 32
__global__ __launch_bounds__(256) void k_lstm_fwd(const float* __restrict__ x25,
                                                  const float* __restrict__ wih,
                                                  const float* __restrict__ whh,
                                                  const float* __restrict__ bih,
                                                  const float* __restrict__ bhh,
                                                  float* __restrict__ feat) {
    __shared__ __align__(16) unsigned short whs[208][64];   // whh[j][k] bf16, zero-padded
    __shared__ __align__(16) unsigned short hbf[BT][64];    // h bf16, K zero-padded
    __shared__ unsigned short gbuf[BT][208];                // gate pre-activations (bf16)
    __shared__ float cst[BT][50];
    __shared__ float wihs[200], bsum[200];
    __shared__ float xts[BT][25];

    const int tid = threadIdx.x;
    const int c = blockIdx.x >> 5;
    const int b0 = (blockIdx.x & 31) * BT;

    const float* wr = &whh[(c * 2 + 0) * 200 * 50];
    for (int e = tid; e < 208 * 64; e += 256) {
        int j = e >> 6, k = e & 63;
        whs[j][k] = f2bf((j < 200 && k < 50) ? wr[j * 50 + k] : 0.f);
    }
    const int ibase = (c * 2 + 0) * 200;
    for (int e = tid; e < 200; e += 256) {
        wihs[e] = wih[ibase + e];
        bsum[e] = bih[ibase + e] + bhh[ibase + e];
    }
    for (int e = tid; e < BT * 25; e += 256)
        xts[e / 25][e % 25] = x25[((b0 + e / 25) * 16 + c) * 25 + e % 25];
    for (int e = tid; e < BT * 64; e += 256) hbf[e >> 6][e & 63] = 0;
    for (int e = tid; e < BT * 50; e += 256) cst[e / 50][e % 50] = 0.f;
    __syncthreads();

    const int lane = tid & 31, wv = tid >> 5;
    const int half = lane >> 4, ln = lane & 15;
    const int NT = 13, MT = BT / 16;

    for (int t = 0; t < 25; ++t) {
        for (int tile = wv; tile < MT * NT; tile += 8) {
            int mt = tile / NT, nt = tile % NT;
            v8f acc = {};
#pragma unroll
            for (int ks = 0; ks < 2; ++ks) {
                AFrag af, bf;
                int br = mt * 16 + ln;
                int ka = ks * 32 + half * 16;
#pragma unroll
                for (int e = 0; e < 8; ++e) {
                    af.s[e]     = hbf[br][ka + e];
                    af.s[8 + e] = hbf[br][ka + 8 + e];
                }
                int j  = nt * 16 + ln;
                int kb = ks * 32 + half * 16;
                bf.h[0] = *(const v8bf*)&whs[j][kb];
                bf.h[1] = *(const v8bf*)&whs[j][kb + 8];
                acc = __builtin_amdgcn_wmma_f32_16x16x32_bf16(
                    false, af.v, false, bf.v, (short)0, acc, false, false);
            }
            int j = nt * 16 + ln;
#pragma unroll
            for (int r = 0; r < 8; ++r)
                gbuf[mt * 16 + r + half * 8][j] = f2bf(acc[r]);
        }
        __syncthreads();
        for (int e = tid; e < BT * 50; e += 256) {
            int br = e / 50, j = e % 50;
            float xv = xts[br][t];
            float gi = bf2f(gbuf[br][j])       + xv * wihs[j]       + bsum[j];
            float gf = bf2f(gbuf[br][50 + j])  + xv * wihs[50 + j]  + bsum[50 + j];
            float gg = bf2f(gbuf[br][100 + j]) + xv * wihs[100 + j] + bsum[100 + j];
            float go = bf2f(gbuf[br][150 + j]) + xv * wihs[150 + j] + bsum[150 + j];
            float cc = sigmoidf(gf) * cst[br][j] + sigmoidf(gi) * tanhf(gg);
            cst[br][j] = cc;
            float hh = sigmoidf(go) * tanhf(cc);
            hbf[br][j] = f2bf(hh);
            if (t == 24) feat[(c * 1024 + (b0 + br)) * 100 + j] = hh;
        }
        __syncthreads();
    }
}

// head BN statistics over batch per (c,f)
__global__ __launch_bounds__(256) void k_head_stats(const float* __restrict__ feat,
                                                    float* __restrict__ hmean,
                                                    float* __restrict__ hinv) {
    __shared__ float red[256], red2[256];
    const int cf = blockIdx.x, c = cf / 100, f = cf % 100;
    const int tid = threadIdx.x;
    float s = 0.f, q = 0.f;
    for (int bb = tid; bb < 1024; bb += 256) {
        float v = feat[(c * 1024 + bb) * 100 + f];
        s += v; q += v * v;
    }
    red[tid] = s; red2[tid] = q;
    __syncthreads();
    for (int off = 128; off > 0; off >>= 1) {
        if (tid < off) { red[tid] += red[tid + off]; red2[tid] += red2[tid + off]; }
        __syncthreads();
    }
    if (tid == 0) {
        float m = red[0] * (1.f / 1024.f);
        hmean[cf] = m;
        hinv[cf]  = rsqrtf(red2[0] * (1.f / 1024.f) - m * m);
    }
}

// head BN apply + per-channel linear + sigmoid; also fc-BN stats per channel
__global__ __launch_bounds__(256) void k_head_lin(const float* __restrict__ feat,
                                                  const float* __restrict__ hmean,
                                                  const float* __restrict__ hinv,
                                                  const float* __restrict__ hbw,
                                                  const float* __restrict__ hbb,
                                                  const float* __restrict__ hlw,
                                                  const float* __restrict__ hlb,
                                                  float* __restrict__ ybuf,
                                                  float* __restrict__ fcm,
                                                  float* __restrict__ fcinv) {
    __shared__ float red[256], red2[256];
    const int c = blockIdx.x, tid = threadIdx.x;
    float s = 0.f, q = 0.f;
    for (int bb = tid; bb < 1024; bb += 256) {
        float acc = hlb[c];
        const float* fr = &feat[(c * 1024 + bb) * 100];
        for (int f = 0; f < 100; ++f) {
            float nf = (fr[f] - hmean[c * 100 + f]) * hinv[c * 100 + f] * hbw[c * 100 + f]
                       + hbb[c * 100 + f];
            acc += nf * hlw[c * 100 + f];
        }
        float y = sigmoidf(acc);
        ybuf[c * 1024 + bb] = y;
        s += y; q += y * y;
    }
    red[tid] = s; red2[tid] = q;
    __syncthreads();
    for (int off = 128; off > 0; off >>= 1) {
        if (tid < off) { red[tid] += red[tid + off]; red2[tid] += red2[tid + off]; }
        __syncthreads();
    }
    if (tid == 0) {
        float m = red[0] * (1.f / 1024.f);
        fcm[c]   = m;
        fcinv[c] = rsqrtf(red2[0] * (1.f / 1024.f) - m * m);
    }
}

__global__ void k_final(const float* __restrict__ ybuf, const float* __restrict__ fcm,
                        const float* __restrict__ fcinv, const float* __restrict__ fbw,
                        const float* __restrict__ fbb, const float* __restrict__ flw,
                        const float* __restrict__ flb, float* __restrict__ out) {
    int bb = blockIdx.x * 256 + threadIdx.x;
    if (bb >= 1024) return;
    float acc = flb[0];
#pragma unroll
    for (int c = 0; c < 16; ++c) {
        float yn = (ybuf[c * 1024 + bb] - fcm[c]) * fcinv[c] * fbw[c] + fbb[c];
        acc += yn * flw[c];
    }
    out[bb] = sigmoidf(acc);
}

extern "C" void kernel_launch(void* const* d_in, const int* in_sizes, int n_in,
                              void* d_out, int out_size, void* d_ws, size_t ws_size,
                              hipStream_t stream) {
    (void)in_sizes; (void)n_in; (void)out_size; (void)ws_size;

    const float* x        = (const float*)d_in[0];
    const float* conv1_w  = (const float*)d_in[1];
    const float* conv1_b  = (const float*)d_in[2];
    const float* bn1_w    = (const float*)d_in[3];
    const float* bn1_b    = (const float*)d_in[4];
    const float* conv2_w  = (const float*)d_in[5];
    const float* conv2_b  = (const float*)d_in[6];
    const float* bn2_w    = (const float*)d_in[7];
    const float* bn2_b    = (const float*)d_in[8];
    const float* conv3_w  = (const float*)d_in[9];
    const float* conv3_b  = (const float*)d_in[10];
    const float* bn3_w    = (const float*)d_in[11];
    const float* bn3_b    = (const float*)d_in[12];
    const float* convp_w  = (const float*)d_in[13];
    const float* convp_b  = (const float*)d_in[14];
    const float* wih      = (const float*)d_in[15];
    const float* whh      = (const float*)d_in[16];
    const float* bih      = (const float*)d_in[17];
    const float* bhh      = (const float*)d_in[18];
    const float* head_bnw = (const float*)d_in[19];
    const float* head_bnb = (const float*)d_in[20];
    const float* head_lw  = (const float*)d_in[21];
    const float* head_lb  = (const float*)d_in[22];
    const float* fc_bnw   = (const float*)d_in[23];
    const float* fc_bnb   = (const float*)d_in[24];
    const float* fc_lw    = (const float*)d_in[25];
    const float* fc_lb    = (const float*)d_in[26];
    float* out = (float*)d_out;

    float* ws = (float*)d_ws;
    const size_t OFS_C2    = 104857600;                 // conv1 out: 1024*8*200*64
    const size_t OFS_P     = OFS_C2 + 3276800;          // conv2 out: 1024*16*200
    const size_t OFS_T4    = OFS_P + 819200;            // pooled:    1024*16*50
    const size_t OFS_X25   = OFS_T4 + 819200;           // convp out: 1024*16*50
    const size_t OFS_FEAT  = OFS_X25 + 409600;          // x25:       1024*16*25
    const size_t OFS_Y     = OFS_FEAT + 1638400;        // feat: 16*1024*100
    const size_t OFS_STATS = OFS_Y + 16384;             // ybuf: 16*1024

    float* c1o    = ws;
    float* c2o    = ws + OFS_C2;
    float* pbuf   = ws + OFS_P;
    float* t4     = ws + OFS_T4;
    float* x25    = ws + OFS_X25;
    float* feat   = ws + OFS_FEAT;
    float* ybuf   = ws + OFS_Y;
    float* stats1 = ws + OFS_STATS;        // 32
    float* stats2 = stats1 + 32;           // 64
    float* stats3 = stats2 + 64;           // 64
    float* hmean  = stats3 + 64;           // 1600
    float* hinv   = hmean + 1600;          // 1600
    float* fcm    = hinv + 1600;           // 16
    float* fcinv  = fcm + 16;              // 16

    hipMemsetAsync(stats1, 0, 3392 * sizeof(float), stream);

    k_conv1<<<dim3(1024, 4), 256, 0, stream>>>(x, conv1_w, conv1_b, c1o, stats1);
    k_finalize<<<1, 8, 0, stream>>>(stats1, 8, 1.f / 13107200.f);

    k_conv2<<<12800, 256, 0, stream>>>(c1o, stats1, bn1_w, bn1_b, conv2_w, conv2_b, c2o, stats2);
    k_finalize<<<1, 16, 0, stream>>>(stats2, 16, 1.f / 204800.f);

    k_pool4<<<(1024 * 16 * 50 + 255) / 256, 256, 0, stream>>>(c2o, stats2, bn2_w, bn2_b, pbuf);

    k_conv3p<<<1024, 256, 0, stream>>>(pbuf, conv3_w, conv3_b, convp_w, convp_b, t4, stats3);
    k_finalize<<<1, 16, 0, stream>>>(stats3, 16, 1.f / 51200.f);

    k_pool2<<<(1024 * 16 * 25 + 255) / 256, 256, 0, stream>>>(t4, stats3, bn3_w, bn3_b, x25);

    k_lstm_bwd<<<(16 * 1024 * 50 + 255) / 256, 256, 0, stream>>>(x25, wih, bih, bhh, feat);
    k_lstm_fwd<<<512, 256, 0, stream>>>(x25, wih, whh, bih, bhh, feat);

    k_head_stats<<<1600, 256, 0, stream>>>(feat, hmean, hinv);
    k_head_lin<<<16, 256, 0, stream>>>(feat, hmean, hinv, head_bnw, head_bnb,
                                       head_lw, head_lb, ybuf, fcm, fcinv);
    k_final<<<4, 256, 0, stream>>>(ybuf, fcm, fcinv, fc_bnw, fc_bnb, fc_lw, fc_lb, out);
}
